// PointsLoss_62457414419096
// MI455X (gfx1250) — compile-verified, matrix operand order checked by reading, not run on hardware
//
#include <hip/hip_runtime.h>
#include <hip/hip_bf16.h>
#include <stdint.h>

// ---------------------------------------------------------------------------
// Problem constants (fixed by the reference's setup_inputs):
//   B=16 batches, C=32 added channels, C1=33 original channels,
//   H=W=256 grid, T=20 boxes, voxel=0.8m.
// ---------------------------------------------------------------------------
#define GRID_H   256
#define NCELL    (GRID_H * GRID_H)   // 65536 cells per batch
#define NB       16
#define NC       32
#define NC1      33
#define NT       20
#define VOX      0.8f
#define INV_VOX  1.25f               // 1/0.8 exactly representable
#define MASK_WORDS (NCELL / 32)      // 2048 uint32 per batch

typedef __attribute__((ext_vector_type(16))) _Float16 v16h;
typedef __attribute__((ext_vector_type(8)))  float    v8f;
typedef __attribute__((ext_vector_type(4)))  float    f4;   // ext-vector float4 (for nontemporal b128 loads)

// ---------------------------------------------------------------------------
// Kernel 0: zero the workspace (in-box bitmasks + per-batch counters).
// ---------------------------------------------------------------------------
__global__ void pl_init_ws(uint32_t* __restrict__ ws, int n) {
    int i = blockIdx.x * blockDim.x + threadIdx.x;
    if (i < n) ws[i] = 0u;
}

// ---------------------------------------------------------------------------
// Kernel 1: rasterize each rotated box into a per-batch bitmask.
// One block per (batch, box). Only cells inside the box's AABB are tested:
// ~0.2M point-in-box tests total instead of 21M in the brute-force form.
// ---------------------------------------------------------------------------
__global__ void pl_raster_boxes(const float* __restrict__ boxes,
                                uint32_t* __restrict__ mask) {
    const int b = blockIdx.x / NT;
    const int t = blockIdx.x % NT;
    const float* bx = boxes + (size_t)(b * NT + t) * 7;
    const float cx = bx[0], cy = bx[1], cz = bx[2];
    const float hx = 0.5f * bx[3], hy = 0.5f * bx[4], hz = 0.5f * bx[5];
    const float yaw = bx[6];
    // z test: point z is the constant 0.8; if the slab misses it, the whole
    // box contributes nothing (reference: |dz_local| <= hz, inclusive).
    if (fabsf(VOX - cz) > hz) return;
    const float c = cosf(yaw), s = sinf(yaw);
    // conservative AABB of the rotated rectangle, padded by 1 cell
    const float ex = fabsf(hx * c) + fabsf(hy * s);
    const float ey = fabsf(hx * s) + fabsf(hy * c);
    const int imin = max(0,          (int)floorf((cx - ex) * INV_VOX) - 1);
    const int imax = min(GRID_H - 1, (int)ceilf ((cx + ex) * INV_VOX) + 1);
    const int jmin = max(0,          (int)floorf((cy - ey) * INV_VOX) - 1);
    const int jmax = min(GRID_H - 1, (int)ceilf ((cy + ey) * INV_VOX) + 1);
    const int w = imax - imin + 1;
    const int h = jmax - jmin + 1;
    if (w <= 0 || h <= 0) return;
    const int n = w * h;
    uint32_t* m = mask + (size_t)b * MASK_WORDS;
    for (int idx = threadIdx.x; idx < n; idx += blockDim.x) {
        const int i = imin + idx % w;
        const int j = jmin + idx / w;
        const float dx = (float)i * VOX - cx;
        const float dy = (float)j * VOX - cy;
        const float rx =  dx * c + dy * s;    // rotate by -yaw (matches ref)
        const float ry = -dx * s + dy * c;
        if (fabsf(rx) < hx && fabsf(ry) < hy) {
            const int cell = i * GRID_H + j;  // ref flattens row-major (i*W+j)
            atomicOr(&m[cell >> 5], 1u << (cell & 31));
        }
    }
}

// ---------------------------------------------------------------------------
// Kernel 2: stream both occupancy tensors once (the ~273 MB that sets the
// 11.7us HBM roofline), 4 cells per thread via b128 nontemporal loads,
// combine with the in-box bitmask, and reduce inter/union per block with a
// single V_WMMA_F32_16X16X32_F16 (A = ones -> D[m][n] = sum_k B[k][n]).
// 512 WMMA B-slots hold exactly the 256 inter + 256 union partials; counts
// are <= 4 per thread so the f16 conversion and f32 accumulation are exact.
// ---------------------------------------------------------------------------
__global__ void __launch_bounds__(256)
pl_stream_iou(const float* __restrict__ added,
              const float* __restrict__ orig,
              const uint32_t* __restrict__ mask,
              float* __restrict__ cnt) {
    const int b = blockIdx.y;
    const int g = blockIdx.x * 256 + threadIdx.x;        // float4 index, 0..16383
    const f4* a4 = (const f4*)added + (size_t)b * NC * (NCELL / 4);
    const f4* o4 = (const f4*)orig  + ((size_t)b * NC1 + 1) * (NCELL / 4); // skip ch 0

    float sa[4] = {0.f, 0.f, 0.f, 0.f};
    float so[4] = {0.f, 0.f, 0.f, 0.f};
    #pragma unroll 8
    for (int c = 0; c < NC; ++c) {
        f4 v = __builtin_nontemporal_load(&a4[(size_t)c * (NCELL / 4) + g]);
        sa[0] += v.x; sa[1] += v.y; sa[2] += v.z; sa[3] += v.w;
    }
    #pragma unroll 8
    for (int c = 0; c < NC1 - 1; ++c) {
        f4 v = __builtin_nontemporal_load(&o4[(size_t)c * (NCELL / 4) + g]);
        so[0] += v.x; so[1] += v.y; so[2] += v.z; so[3] += v.w;
    }

    const int cell0 = g * 4;
    const uint32_t mw = mask[(size_t)b * MASK_WORDS + (cell0 >> 5)];
    const int sh = cell0 & 31;
    int inter = 0, uni = 0;
    #pragma unroll
    for (int k = 0; k < 4; ++k) {
        const bool ib = (mw >> (sh + k)) & 1u;
        const bool p  = ib && (sa[k] != 0.0f);   // values are >= 0: sum!=0 <=> any nonzero
        const bool o  = ib && (so[k] != 0.0f);
        inter += (p && o) ? 1 : 0;
        uni   += (p || o) ? 1 : 0;
    }

    __shared__ float s_i[256];
    __shared__ float s_u[256];
    __shared__ float s_red[16];
    s_i[threadIdx.x] = (float)inter;
    s_u[threadIdx.x] = (float)uni;
    __syncthreads();

    if (threadIdx.x < 32) {                    // entire wave 0: EXEC all-ones
        const int lane = threadIdx.x;
        const int n = lane & 15;               // WMMA column (N = lane%16)
        const int h = lane >> 4;
        v16h aones, bfrag;
        #pragma unroll
        for (int i = 0; i < 16; ++i) aones[i] = (_Float16)1.0f;
        // columns 0..7 <- inter partials, columns 8..15 <- union partials;
        // each column's two lanes x 16 slots cover 32 distinct partials.
        const float* src = (n < 8) ? (s_i + n * 32 + h * 16)
                                   : (s_u + (n - 8) * 32 + h * 16);
        #pragma unroll
        for (int j = 0; j < 16; ++j) bfrag[j] = (_Float16)src[j];
        v8f accv = {};
        accv = __builtin_amdgcn_wmma_f32_16x16x32_f16(
            /*neg_a=*/false, aones, /*neg_b=*/false, bfrag,
            /*c_mod=*/(short)0, accv, /*reuse_a=*/false, /*reuse_b=*/false);
        // All D rows are identical; VGPR0 on lanes 0..15 holds column n's sum.
        if (lane < 16) s_red[lane] = accv[0];
    }
    __syncthreads();

    if (threadIdx.x == 0) {
        float ti = 0.f, tu = 0.f;
        #pragma unroll
        for (int i = 0; i < 8; ++i) { ti += s_red[i]; tu += s_red[8 + i]; }
        atomicAdd(&cnt[2 * b + 0], ti);
        atomicAdd(&cnt[2 * b + 1], tu);
    }
}

// ---------------------------------------------------------------------------
// Kernel 3: scalar finalize:  out = T * sum_b( inter_b / max(union_b,1) ) / B
// ---------------------------------------------------------------------------
__global__ void pl_finalize(const float* __restrict__ cnt, float* __restrict__ out) {
    if (blockIdx.x == 0 && threadIdx.x == 0) {
        float s = 0.0f;
        #pragma unroll
        for (int b = 0; b < NB; ++b)
            s += cnt[2 * b + 0] / fmaxf(cnt[2 * b + 1], 1.0f);
        out[0] = (float)NT * s / (float)NB;
    }
}

// ---------------------------------------------------------------------------
extern "C" void kernel_launch(void* const* d_in, const int* in_sizes, int n_in,
                              void* d_out, int out_size, void* d_ws, size_t ws_size,
                              hipStream_t stream) {
    const float* added = (const float*)d_in[0];  // (16,32,256,256) f32
    const float* orig  = (const float*)d_in[1];  // (16,33,256,256) f32
    const float* boxes = (const float*)d_in[2];  // (16,20,7) f32
    // d_in[3] (tf_ego) is unused by the reference.

    uint32_t* mask = (uint32_t*)d_ws;                               // 16*2048 words
    float*    cnt  = (float*)((char*)d_ws + (size_t)NB * MASK_WORDS * 4); // 16*(inter,union)

    const int ninit = NB * MASK_WORDS + 2 * NB;
    pl_init_ws<<<(ninit + 255) / 256, 256, 0, stream>>>((uint32_t*)d_ws, ninit);
    pl_raster_boxes<<<NB * NT, 128, 0, stream>>>(boxes, mask);
    pl_stream_iou<<<dim3(NCELL / 4 / 256, NB), 256, 0, stream>>>(added, orig, mask, cnt);
    pl_finalize<<<1, 32, 0, stream>>>(cnt, (float*)d_out);
}
